// GCN2_LSTM_89008902243172
// MI455X (gfx1250) — matrix operand order repaired and checked by reading, not compiled, not run on hardware
//
#include <hip/hip_runtime.h>

// ---- problem constants (reference: N=100000 divisible by 16, F=64, S=64, C=32) ----
constexpr int NF = 64;   // node feature / hidden dim
constexpr int NC = 32;   // num classes

typedef __attribute__((ext_vector_type(16))) __bf16 v16bf;
typedef __attribute__((ext_vector_type(8)))  float  v8f;

__device__ __forceinline__ v8f wmma_bf16(v16bf a, v16bf b, v8f c) {
  return __builtin_amdgcn_wmma_f32_16x16x32_bf16(false, a, false, b, (short)0, c,
                                                 false, false);
}

// ---- WMMA fragment loaders (contiguous K order shared by A and B, wave32) -------------------
// A: 16x32 (MxK) bf16. lane&15 = M row; lane>>4 selects K 0-15 vs 16-31.
// VGPR v: K = kb+2v,kb+2v+1.
__device__ __forceinline__ v16bf load_a_f32(const float* base, int ld, int m0,
                                            int k0, int lane) {
  int m  = m0 + (lane & 15);
  int kb = k0 + ((lane >> 4) << 4);
  const float* p = base + m * ld + kb;
  v16bf a;
#pragma unroll
  for (int v = 0; v < 8; ++v) {
    int kk = 2 * v;
    a[2 * v]     = (__bf16)p[kk];
    a[2 * v + 1] = (__bf16)p[kk + 1];
  }
  return a;
}

__device__ __forceinline__ v16bf load_a_bf(const __bf16* base, int ld, int m0,
                                           int k0, int lane) {
  int m  = m0 + (lane & 15);
  int kb = k0 + ((lane >> 4) << 4);
  const __bf16* p = base + m * ld + kb;
  v16bf a;
#pragma unroll
  for (int v = 0; v < 8; ++v) {
    int kk = 2 * v;
    a[2 * v]     = p[kk];
    a[2 * v + 1] = p[kk + 1];
  }
  return a;
}

// B: 32x16 (KxN). lane&15 = N col; lane>>4 selects K 0-15 vs 16-31; VGPR v: K=kb+2v(+1).
// Row-major W[K x OUT]:
__device__ __forceinline__ v16bf load_b_rm(const float* W, int ldw, int k0, int n0,
                                           int lane) {
  int n  = n0 + (lane & 15);
  int kb = k0 + ((lane >> 4) << 4);
  v16bf b;
#pragma unroll
  for (int v = 0; v < 8; ++v) {
    b[2 * v]     = (__bf16)W[(kb + 2 * v) * ldw + n];
    b[2 * v + 1] = (__bf16)W[(kb + 2 * v + 1) * ldw + n];
  }
  return b;
}

// Transposed weight (W stored [OUTxK] row-major; B[k][n] = W[n][k]) -- LSTM W_ih.
__device__ __forceinline__ v16bf load_b_tr(const float* W, int ldw, int k0, int n0,
                                           int lane) {
  int n  = n0 + (lane & 15);
  int kb = k0 + ((lane >> 4) << 4);
  const float* p = W + n * ldw + kb;
  v16bf b;
#pragma unroll
  for (int v = 0; v < 8; ++v) {
    b[2 * v]     = (__bf16)p[2 * v];
    b[2 * v + 1] = (__bf16)p[2 * v + 1];
  }
  return b;
}

__device__ __forceinline__ float sigmoidf_dev(float x) {
  return 1.0f / (1.0f + __expf(-x));
}

// ---- 1) fused LSTM step: gates = x @ W_ih^T (+b_ih+b_hh), h = sig(o)*tanh(sig(i)*tanh(g))
// block = 128 threads (4 waves), 16 rows per block, each wave covers one 64-col gate.
__global__ __launch_bounds__(128) void lstm_kernel(const float* __restrict__ x,
                                                   const float* __restrict__ W_ih,
                                                   const float* __restrict__ b_ih,
                                                   const float* __restrict__ b_hh,
                                                   __bf16* __restrict__ h_bf) {
  __shared__ float g_lds[16 * 256];
  const int lane = threadIdx.x & 31;
  const int wave = threadIdx.x >> 5;
  const int m0   = blockIdx.x * 16;

  v16bf a0 = load_a_f32(x, NF, m0, 0, lane);
  v16bf a1 = load_a_f32(x, NF, m0, 32, lane);

#pragma unroll
  for (int t = 0; t < 4; ++t) {
    const int n0 = (wave * 4 + t) * 16;
    v16bf b0 = load_b_tr(W_ih, NF, 0, n0, lane);
    v16bf b1 = load_b_tr(W_ih, NF, 32, n0, lane);
    v8f acc = {};
    acc = wmma_bf16(a0, b0, acc);
    acc = wmma_bf16(a1, b1, acc);
    const int ncol = lane & 15;
    const int mofs = (lane >> 4) * 8;
#pragma unroll
    for (int v = 0; v < 8; ++v)
      g_lds[(mofs + v) * 256 + n0 + ncol] = acc[v];
  }
  __syncthreads();

  for (int idx = threadIdx.x; idx < 16 * NF; idx += blockDim.x) {
    const int m = idx >> 6;
    const int j = idx & 63;
    const float gi = g_lds[m * 256 + j]        + b_ih[j]        + b_hh[j];
    const float gg = g_lds[m * 256 + 128 + j]  + b_ih[128 + j]  + b_hh[128 + j];
    const float go = g_lds[m * 256 + 192 + j]  + b_ih[192 + j]  + b_hh[192 + j];
    const float c  = sigmoidf_dev(gi) * tanhf(gg);
    const float h  = sigmoidf_dev(go) * tanhf(c);
    h_bf[(m0 + m) * NF + j] = (__bf16)h;
  }
}

// ---- generic bf16 WMMA GEMM: out[N x OUT] = in[N x 64] @ W[64 x OUT] -----------
template <int OUT>
__global__ __launch_bounds__(256) void gemm_bf16(const __bf16* __restrict__ in,
                                                 const float* __restrict__ W,
                                                 float* __restrict__ out,
                                                 int totalTiles) {
  const int lane  = threadIdx.x & 31;
  const int wtile = blockIdx.x * (blockDim.x >> 5) + (threadIdx.x >> 5);
  if (wtile >= totalTiles) return;  // wave-uniform: EXEC stays all-ones for WMMA
  constexpr int CT = OUT / 16;
  const int m0 = (wtile / CT) * 16;
  const int n0 = (wtile % CT) * 16;

  v16bf a0 = load_a_bf(in, NF, m0, 0, lane);
  v16bf a1 = load_a_bf(in, NF, m0, 32, lane);
  v16bf b0 = load_b_rm(W, OUT, 0, n0, lane);
  v16bf b1 = load_b_rm(W, OUT, 32, n0, lane);
  v8f acc = {};
  acc = wmma_bf16(a0, b0, acc);
  acc = wmma_bf16(a1, b1, acc);

  const int ncol = lane & 15;
  const int mofs = (lane >> 4) * 8;
#pragma unroll
  for (int v = 0; v < 8; ++v)
    out[(m0 + mofs + v) * OUT + n0 + ncol] = acc[v];
}

// ---- degree / normalization ----------------------------------------------------
__global__ void deg_init(float* deg, int n) {
  int i = blockIdx.x * blockDim.x + threadIdx.x;
  if (i < n) deg[i] = 1.0f;  // self-loop
}
__global__ void deg_count(const int* __restrict__ col, float* deg, int e) {
  int i = blockIdx.x * blockDim.x + threadIdx.x;
  if (i < e) atomicAdd(&deg[col[i]], 1.0f);
}
__global__ void dinv_kernel(const float* __restrict__ deg, float* dinv, int n) {
  int i = blockIdx.x * blockDim.x + threadIdx.x;
  if (i < n) dinv[i] = __frsqrt_rn(deg[i]);  // deg >= 1 always
}

// ---- aggregation: init with bias + self-loop term, then atomic edge scatter ----
__global__ void agg_init(const float* __restrict__ hw, const float* __restrict__ bias,
                         const float* __restrict__ dinv, float* __restrict__ out,
                         int n, int outDim) {
  int t = blockIdx.x * blockDim.x + threadIdx.x;
  if (t >= n * outDim) return;
  int i = t / outDim;
  int f = t - i * outDim;
  float di = dinv[i];
  out[t] = bias[f] + hw[t] * di * di;
}

template <int OUT, int SHIFT>
__global__ void scatter_kernel(const float* __restrict__ hw,
                               const int* __restrict__ row,
                               const int* __restrict__ col,
                               const float* __restrict__ dinv,
                               float* __restrict__ out, int total) {
  int t = blockIdx.x * blockDim.x + threadIdx.x;
  if (t >= total) return;
  int e = t >> SHIFT;
  int f = t & (OUT - 1);
  int r = row[e], c = col[e];
  float w = dinv[r] * dinv[c];
  atomicAdd(out + c * OUT + f, hw[r * OUT + f] * w);
}

__global__ void relu_to_bf(const float* __restrict__ in, __bf16* __restrict__ out,
                           int total) {
  int t = blockIdx.x * blockDim.x + threadIdx.x;
  if (t < total) out[t] = (__bf16)fmaxf(in[t], 0.0f);
}

// ---- link decode: dot(z[src], z[dst]) over 32 dims -----------------------------
__global__ void decode_kernel(const float* __restrict__ z, const int* __restrict__ eli,
                              float* __restrict__ out, int el) {
  int e = blockIdx.x * blockDim.x + threadIdx.x;
  if (e >= el) return;
  int s = eli[e];
  int d = eli[el + e];
  const float4* zs = (const float4*)(z + s * NC);
  const float4* zd = (const float4*)(z + d * NC);
  float acc = 0.0f;
#pragma unroll
  for (int i = 0; i < NC / 4; ++i) {
    float4 a = zs[i], b = zd[i];
    acc += a.x * b.x + a.y * b.y + a.z * b.z + a.w * b.w;
  }
  out[e] = acc;
}

extern "C" void kernel_launch(void* const* d_in, const int* in_sizes, int n_in,
                              void* d_out, int out_size, void* d_ws, size_t ws_size,
                              hipStream_t stream) {
  const float* x    = (const float*)d_in[0];
  const int*   ei   = (const int*)d_in[1];
  const int*   eli  = (const int*)d_in[2];
  const float* W_ih = (const float*)d_in[3];
  // d_in[4] = W_hh: mathematically dead (h0 == 0)
  const float* b_ih = (const float*)d_in[5];
  const float* b_hh = (const float*)d_in[6];
  const float* W1   = (const float*)d_in[7];
  const float* b1   = (const float*)d_in[8];
  const float* W2   = (const float*)d_in[9];
  const float* b2   = (const float*)d_in[10];

  const int N  = in_sizes[0] / NF;  // 100000 (divisible by 16)
  const int E  = in_sizes[1] / 2;   // 1600000
  const int EL = in_sizes[2] / 2;   // 1000000
  const int* row = ei;
  const int* col = ei + E;

  // workspace layout
  char* ws = (char*)d_ws;
  size_t off = 0;
  auto take = [&](size_t bytes) -> char* {
    char* p = ws + off;
    off = (off + bytes + 255) & ~(size_t)255;
    return p;
  };
  __bf16* h_bf  = (__bf16*)take((size_t)N * NF * sizeof(__bf16));
  float*  hw1   = (float*) take((size_t)N * NF * sizeof(float));
  float*  out1  = (float*) take((size_t)N * NF * sizeof(float));
  __bf16* h2_bf = (__bf16*)take((size_t)N * NF * sizeof(__bf16));
  float*  hw2   = (float*) take((size_t)N * NC * sizeof(float));
  float*  z     = (float*) take((size_t)N * NC * sizeof(float));
  float*  deg   = (float*) take((size_t)N * sizeof(float));
  float*  dinv  = (float*) take((size_t)N * sizeof(float));

  const int rowTiles = N / 16;

  // 1) LSTM step (WMMA GEMM + fused activations)
  lstm_kernel<<<rowTiles, 128, 0, stream>>>(x, W_ih, b_ih, b_hh, h_bf);

  // 2) degree + normalization (re-initialized every call for graph replay)
  deg_init<<<(N + 255) / 256, 256, 0, stream>>>(deg, N);
  deg_count<<<(E + 255) / 256, 256, 0, stream>>>(col, deg, E);
  dinv_kernel<<<(N + 255) / 256, 256, 0, stream>>>(deg, dinv, N);

  // 3) conv1: hw1 = h @ W1 ; aggregate ; relu -> bf16
  {
    int tiles = rowTiles * (NF / 16);
    gemm_bf16<NF><<<(tiles + 7) / 8, 256, 0, stream>>>(h_bf, W1, hw1, tiles);
  }
  agg_init<<<(N * NF + 255) / 256, 256, 0, stream>>>(hw1, b1, dinv, out1, N, NF);
  {
    int total = E * NF;
    scatter_kernel<NF, 6><<<(total + 255) / 256, 256, 0, stream>>>(hw1, row, col,
                                                                   dinv, out1, total);
  }
  relu_to_bf<<<(N * NF + 255) / 256, 256, 0, stream>>>(out1, h2_bf, N * NF);

  // 4) conv2: hw2 = h2 @ W2 ; aggregate -> z
  {
    int tiles = rowTiles * (NC / 16);
    gemm_bf16<NC><<<(tiles + 7) / 8, 256, 0, stream>>>(h2_bf, W2, hw2, tiles);
  }
  agg_init<<<(N * NC + 255) / 256, 256, 0, stream>>>(hw2, b2, dinv, z, N, NC);
  {
    int total = E * NC;
    scatter_kernel<NC, 5><<<(total + 255) / 256, 256, 0, stream>>>(hw2, row, col,
                                                                   dinv, z, total);
  }

  // 5) link decode
  decode_kernel<<<(EL + 255) / 256, 256, 0, stream>>>(z, eli, (float*)d_out, EL);
}
